// SimplifiedMoE_84928683311860
// MI455X (gfx1250) — matrix operand order repaired and checked by reading, not compiled, not run on hardware
//
#include <hip/hip_runtime.h>

// Problem constants (from reference): B=4, T=2048, E=8, C=1024, I=2048
#define BT_   8192
#define E_    8
#define C_    1024
#define I_    2048

#define MTILE 32              // rows of (B*T) per workgroup
#define LDH   (I_ + 8)        // padded LDS row stride in ushorts (4112B, 16B-aligned)

typedef __attribute__((ext_vector_type(16))) __bf16        v16bf;
typedef __attribute__((ext_vector_type(8)))  float         v8f;
typedef __attribute__((ext_vector_type(8)))  unsigned int  v8u;   // 32B
typedef __attribute__((ext_vector_type(4)))  unsigned int  v4u;   // 16B

__device__ __forceinline__ unsigned short bf16_bits(float f) {
  unsigned u = __builtin_bit_cast(unsigned, f);
  u += 0x7FFFu + ((u >> 16) & 1u);                  // round-to-nearest-even
  return (unsigned short)(u >> 16);
}

__device__ __forceinline__ __bf16 bf16_rne(float f) {
  return __builtin_bit_cast(__bf16, bf16_bits(f));
}

__device__ __forceinline__ float gelu_erf(float x) {
  return 0.5f * x * (1.0f + erff(x * 0.70710678118654752440f));
}

// ---- fragment builders -----------------------------------------------------
// B fragment (pre-converted bf16): lane = N column, K rows sel*8+j and 16+sel*8+j (stride ld)
__device__ __forceinline__ v16bf load_frag_contig_bf16(const unsigned short* p, int ld, int sel) {
  v16bf b;
  #pragma unroll
  for (int j = 0; j < 8; ++j) {
    b[j]     = __builtin_bit_cast(__bf16, p[(size_t)(sel * 8 + j) * ld]);
    b[8 + j] = __builtin_bit_cast(__bf16, p[(size_t)(16 + sel * 8 + j) * ld]);
  }
  return b;
}
// A fragment (bf16, global or LDS): two 16B runs at sel*8 and 16+sel*8
__device__ __forceinline__ v16bf load_frag_split_bf16(const unsigned short* p, int sel) {
  v4u lo = *(const v4u*)(p + sel * 8);
  v4u hi = *(const v4u*)(p + 16 + sel * 8);
  return __builtin_bit_cast(v16bf, __builtin_shufflevector(lo, hi, 0, 1, 2, 3, 4, 5, 6, 7));
}
// fallback (fp32 source, convert in-loop)
__device__ __forceinline__ v16bf load_frag_contig_f32(const float* p, int ld, int sel) {
  v16bf b;
  #pragma unroll
  for (int j = 0; j < 8; ++j) {
    b[j]     = bf16_rne(p[(size_t)(sel * 8 + j) * ld]);
    b[8 + j] = bf16_rne(p[(size_t)(16 + sel * 8 + j) * ld]);
  }
  return b;
}
__device__ __forceinline__ v16bf load_frag_split_f32(const float* p, int sel) {
  v16bf a;
  #pragma unroll
  for (int j = 0; j < 8; ++j) {
    a[j]     = bf16_rne(p[sel * 8 + j]);
    a[8 + j] = bf16_rne(p[16 + sel * 8 + j]);
  }
  return a;
}

// ---------------------------------------------------------------------------
// Preconvert: fp32 -> bf16, 8 elements/thread (float4 x2 in, uint4 out)
// ---------------------------------------------------------------------------
__global__ __launch_bounds__(256)
void convert_f32_bf16_kernel(const float* __restrict__ src,
                             unsigned short* __restrict__ dst, int n8) {
  int i = blockIdx.x * blockDim.x + threadIdx.x;
  if (i >= n8) return;
  const float4* s4 = (const float4*)src;
  float4 f0 = s4[2 * i], f1 = s4[2 * i + 1];
  uint4 o;
  o.x = (unsigned)bf16_bits(f0.x) | ((unsigned)bf16_bits(f0.y) << 16);
  o.y = (unsigned)bf16_bits(f0.z) | ((unsigned)bf16_bits(f0.w) << 16);
  o.z = (unsigned)bf16_bits(f1.x) | ((unsigned)bf16_bits(f1.y) << 16);
  o.w = (unsigned)bf16_bits(f1.z) | ((unsigned)bf16_bits(f1.w) << 16);
  ((uint4*)dst)[i] = o;
}

// ---------------------------------------------------------------------------
// Kernel 1: per (expert e, 32-row tile):
//   h  = gelu(x[m0:m0+32,:] @ w1[e])   (bf16, kept in LDS, 32 x 2048)
//   eo = h @ w2[e]                      (fp32 -> experts_out)
// 256 threads = 8 waves, v_wmma_f32_16x16x32_bf16.
// PRE=true: operands pre-converted to bf16 in workspace (no VALU in K-loop).
// ---------------------------------------------------------------------------
template <bool PRE>
__global__ __launch_bounds__(256)
void moe_expert_kernel(const float* __restrict__ x,
                       const float* __restrict__ w1,
                       const float* __restrict__ w2,
                       const unsigned short* __restrict__ xb,
                       const unsigned short* __restrict__ w1b,
                       const unsigned short* __restrict__ w2b,
                       float* __restrict__ experts_out) {
  extern __shared__ unsigned short h_s[];          // [MTILE][LDH] bf16 bits

  const int e    = blockIdx.y;
  const int m0   = blockIdx.x * MTILE;
  const int lane = threadIdx.x & 31;
  const int wave = threadIdx.x >> 5;
  const int sel  = lane >> 4;                      // 0: lanes 0-15, 1: lanes 16-31
  const int l15  = lane & 15;

  const float*          w1e  = w1  + (size_t)e * C_ * I_;   // [C][I]
  const float*          w2e  = w2  + (size_t)e * I_ * C_;   // [I][C]
  const unsigned short* w1eb = w1b + (size_t)e * C_ * I_;
  const unsigned short* w2eb = w2b + (size_t)e * I_ * C_;

  // ---------------- Phase 1: h = gelu(x @ w1[e]) ----------------
  for (int chunk = 0; chunk < 4; ++chunk) {
    const int n0 = wave * 256 + chunk * 64;
    v8f acc[2][4];
    #pragma unroll
    for (int mt = 0; mt < 2; ++mt)
      #pragma unroll
      for (int nt = 0; nt < 4; ++nt)
        acc[mt][nt] = (v8f)(0.0f);

    for (int kk = 0; kk < C_; kk += 32) {
      v16bf a[2];
      #pragma unroll
      for (int mt = 0; mt < 2; ++mt) {
        const size_t row = (size_t)(m0 + mt * 16 + l15);
        a[mt] = PRE ? load_frag_split_bf16(xb + row * C_ + kk, sel)
                    : load_frag_split_f32 (x  + row * C_ + kk, sel);
      }
      #pragma unroll
      for (int nt = 0; nt < 4; ++nt) {
        const size_t boff = (size_t)kk * I_ + (n0 + nt * 16 + l15);
        v16bf b = PRE ? load_frag_contig_bf16(w1eb + boff, I_, sel)
                      : load_frag_contig_f32 (w1e  + boff, I_, sel);
        #pragma unroll
        for (int mt = 0; mt < 2; ++mt)
          acc[mt][nt] = __builtin_amdgcn_wmma_f32_16x16x32_bf16(
              false, a[mt], false, b, (short)0, acc[mt][nt], false, false);
      }
    }

    // exact-erf GELU, park bf16 tile in LDS
    #pragma unroll
    for (int mt = 0; mt < 2; ++mt)
      #pragma unroll
      for (int nt = 0; nt < 4; ++nt)
        #pragma unroll
        for (int r = 0; r < 8; ++r) {
          float g = gelu_erf(acc[mt][nt][r]);
          h_s[(mt * 16 + sel * 8 + r) * LDH + (n0 + nt * 16 + l15)] = bf16_bits(g);
        }
  }

  __syncthreads();

  // ---------------- Phase 2: experts_out = h @ w2[e] ----------------
  for (int chunk = 0; chunk < 2; ++chunk) {
    const int n0 = wave * 128 + chunk * 64;
    v8f acc[2][4];
    #pragma unroll
    for (int mt = 0; mt < 2; ++mt)
      #pragma unroll
      for (int nt = 0; nt < 4; ++nt)
        acc[mt][nt] = (v8f)(0.0f);

    for (int kk = 0; kk < I_; kk += 32) {
      if (kk + 32 < I_) {
        if (PRE) __builtin_prefetch(w2eb + (size_t)(kk + 32 + lane) * C_ + n0, 0, 3);
        else     __builtin_prefetch(w2e  + (size_t)(kk + 32 + lane) * C_ + n0, 0, 3);
      }
      v16bf a[2];
      #pragma unroll
      for (int mt = 0; mt < 2; ++mt)
        a[mt] = load_frag_split_bf16(&h_s[(size_t)(mt * 16 + l15) * LDH + kk], sel);

      #pragma unroll
      for (int nt = 0; nt < 4; ++nt) {
        const size_t boff = (size_t)kk * C_ + (n0 + nt * 16 + l15);
        v16bf b = PRE ? load_frag_contig_bf16(w2eb + boff, C_, sel)
                      : load_frag_contig_f32 (w2e  + boff, C_, sel);
        #pragma unroll
        for (int mt = 0; mt < 2; ++mt)
          acc[mt][nt] = __builtin_amdgcn_wmma_f32_16x16x32_bf16(
              false, a[mt], false, b, (short)0, acc[mt][nt], false, false);
      }
    }

    // store fp32 experts_out[bt][e][c]
    #pragma unroll
    for (int mt = 0; mt < 2; ++mt)
      #pragma unroll
      for (int nt = 0; nt < 4; ++nt)
        #pragma unroll
        for (int r = 0; r < 8; ++r) {
          int m = m0 + mt * 16 + sel * 8 + r;
          int n = n0 + nt * 16 + l15;
          experts_out[((size_t)m * E_ + e) * C_ + n] = acc[mt][nt][r];
        }
  }
}

// ---------------------------------------------------------------------------
// Kernel 2: agg[bt][c] = sum_e experts_out[bt][e][c] * rw[bt][e]
// ---------------------------------------------------------------------------
__global__ __launch_bounds__(256)
void moe_combine_kernel(const float* __restrict__ experts_out,
                        const float* __restrict__ rw,
                        float* __restrict__ agg) {
  const size_t idx4 = (size_t)blockIdx.x * blockDim.x + threadIdx.x;  // float4 units
  const size_t bt   = idx4 / (C_ / 4);
  const int    c4   = (int)(idx4 % (C_ / 4));
  const float4* eo  = (const float4*)experts_out;

  float4 s = make_float4(0.f, 0.f, 0.f, 0.f);
  #pragma unroll
  for (int e = 0; e < E_; ++e) {
    float  w = rw[bt * E_ + e];
    float4 v = eo[(bt * E_ + e) * (C_ / 4) + c4];
    s.x += w * v.x; s.y += w * v.y; s.z += w * v.z; s.w += w * v.w;
  }
  ((float4*)agg)[idx4] = s;
}

extern "C" void kernel_launch(void* const* d_in, const int* in_sizes, int n_in,
                              void* d_out, int out_size, void* d_ws, size_t ws_size,
                              hipStream_t stream) {
  (void)in_sizes; (void)n_in; (void)out_size;

  const float* x  = (const float*)d_in[0];   // [B,T,C]
  const float* rw = (const float*)d_in[1];   // [B,T,E]
  const float* w1 = (const float*)d_in[2];   // [E,C,I]
  const float* w2 = (const float*)d_in[3];   // [E,I,C]

  float* agg         = (float*)d_out;                      // [B,T,C]
  float* experts_out = agg + (size_t)BT_ * C_;             // [B,T,E,C]

  const size_t n_x  = (size_t)BT_ * C_;                    //  8,388,608
  const size_t n_w1 = (size_t)E_ * C_ * I_;                // 16,777,216
  const size_t n_w2 = (size_t)E_ * I_ * C_;                // 16,777,216
  const size_t ws_needed = (n_x + n_w1 + n_w2) * sizeof(unsigned short); // ~84 MB

  const size_t lds_bytes = (size_t)MTILE * LDH * sizeof(unsigned short); // 128.5 KB
  dim3 g1(BT_ / MTILE, E_);

  if (ws_size >= ws_needed) {
    unsigned short* xb  = (unsigned short*)d_ws;
    unsigned short* w1b = xb + n_x;
    unsigned short* w2b = w1b + n_w1;

    convert_f32_bf16_kernel<<<(int)(n_x  / 8 / 256), 256, 0, stream>>>(x,  xb,  (int)(n_x  / 8));
    convert_f32_bf16_kernel<<<(int)(n_w1 / 8 / 256), 256, 0, stream>>>(w1, w1b, (int)(n_w1 / 8));
    convert_f32_bf16_kernel<<<(int)(n_w2 / 8 / 256), 256, 0, stream>>>(w2, w2b, (int)(n_w2 / 8));

    moe_expert_kernel<true><<<g1, 256, lds_bytes, stream>>>(
        x, w1, w2, xb, w1b, w2b, experts_out);
  } else {
    // fallback: convert in-loop (no workspace assumption)
    moe_expert_kernel<false><<<g1, 256, lds_bytes, stream>>>(
        x, w1, w2, nullptr, nullptr, nullptr, experts_out);
  }

  const int n4 = BT_ * C_ / 4;
  moe_combine_kernel<<<n4 / 256, 256, 0, stream>>>(experts_out, rw, agg);
}
